// TransformerLayer_2869038154392
// MI455X (gfx1250) — matrix operand order repaired and checked by reading, compiled, not run on hardware
//
#include <hip/hip_runtime.h>

typedef __bf16 bf16;
typedef __attribute__((ext_vector_type(8)))  float  f32x8;
typedef __attribute__((ext_vector_type(8)))  __bf16 bf16x8;
typedef __attribute__((ext_vector_type(16))) __bf16 bf16x16;

union ABfrag { bf16x16 v; bf16x8 h[2]; };

#define WMMA_BF16(a, b, c) __builtin_amdgcn_wmma_f32_16x16x32_bf16(false, (a), false, (b), (short)0, (c), false, false)

static const int S_DIM = 4096;
static const int D_DIM = 2048;
static const int H_DIM = 4;
static const int FFN   = 512;

// ---------------------------------------------------------------------------
// Elementwise f32 -> bf16 convert
// ---------------------------------------------------------------------------
__global__ void convert_bf16_kernel(const float* __restrict__ src,
                                    bf16* __restrict__ dst, int n) {
  int i = blockIdx.x * blockDim.x + threadIdx.x;
  if (i < n) dst[i] = (bf16)src[i];
}

// ---------------------------------------------------------------------------
// Batched transpose+convert: src [B][R][C] f32 -> dst [B][C][R] bf16
// ---------------------------------------------------------------------------
__global__ void transpose_bf16_kernel(const float* __restrict__ src,
                                      bf16* __restrict__ dst,
                                      int B, int R, int C) {
  int i = blockIdx.x * blockDim.x + threadIdx.x;
  int total = B * R * C;
  if (i >= total) return;
  int b = i / (R * C);
  int rem = i - b * (R * C);
  int c = rem / R;
  int r = rem - c * R;
  dst[i] = (bf16)src[(size_t)b * R * C + (size_t)r * C + c];
}

// ---------------------------------------------------------------------------
// QKV projection: q,k [H,S,32] row-major bf16; v stored transposed [H,32,S].
// One wave per (32-row tile, head): 2 A row-tiles x 6 weight tiles = 12 WMMA
// accumulators; each weight B-frag amortized over both row tiles.
// ---------------------------------------------------------------------------
__global__ __launch_bounds__(32) void qkv_kernel(
    const bf16* __restrict__ xh,   // [S, D]
    const bf16* __restrict__ wqT,  // [H, 32, D]
    const bf16* __restrict__ wkT,  // [H, 32, D]
    const bf16* __restrict__ wvT,  // [H, 32, D]
    bf16* __restrict__ qh,         // [H, S, 32]
    bf16* __restrict__ kh,         // [H, S, 32]
    bf16* __restrict__ vT) {       // [H, 32, S]
  const int lane = threadIdx.x;
  const int l15 = lane & 15;
  const int hg = lane >> 4;
  const int rb = blockIdx.x * 32;
  const int h = blockIdx.y;

  const bf16* arow0 = xh + (size_t)(rb + l15) * D_DIM + hg * 16;
  const bf16* arow1 = xh + (size_t)(rb + 16 + l15) * D_DIM + hg * 16;
  const bf16* wq = wqT + (size_t)h * 32 * D_DIM;
  const bf16* wk = wkT + (size_t)h * 32 * D_DIM;
  const bf16* wv = wvT + (size_t)h * 32 * D_DIM;

  // acc[t][m][g]: t = row tile, m = 0:q 1:k 2:v, g = n-group
  f32x8 acc[2][3][2];
#pragma unroll
  for (int t = 0; t < 2; ++t)
#pragma unroll
    for (int m = 0; m < 3; ++m)
#pragma unroll
      for (int g = 0; g < 2; ++g) acc[t][m][g] = (f32x8){};

  for (int kb = 0; kb < D_DIM; kb += 32) {
    ABfrag a0, a1;
    a0.h[0] = *(const bf16x8*)(arow0 + kb);
    a0.h[1] = *(const bf16x8*)(arow0 + kb + 8);
    a1.h[0] = *(const bf16x8*)(arow1 + kb);
    a1.h[1] = *(const bf16x8*)(arow1 + kb + 8);
    ABfrag b;
#define DO_MAT(W, m, g)                                                    \
  {                                                                        \
    const bf16* bc = (W) + (size_t)((g)*16 + l15) * D_DIM + hg * 16 + kb;  \
    b.h[0] = *(const bf16x8*)bc;                                           \
    b.h[1] = *(const bf16x8*)(bc + 8);                                     \
    acc[0][m][g] = WMMA_BF16(a0.v, b.v, acc[0][m][g]);                     \
    acc[1][m][g] = WMMA_BF16(a1.v, b.v, acc[1][m][g]);                     \
  }
    DO_MAT(wq, 0, 0); DO_MAT(wq, 0, 1);
    DO_MAT(wk, 1, 0); DO_MAT(wk, 1, 1);
    DO_MAT(wv, 2, 0); DO_MAT(wv, 2, 1);
#undef DO_MAT
  }

  // q, k: row-major stores.  D-tile: lane holds col n=l15, rows r + 8*hg.
  bf16* qdst = qh + (size_t)h * S_DIM * 32;
  bf16* kdst = kh + (size_t)h * S_DIM * 32;
#pragma unroll
  for (int t = 0; t < 2; ++t) {
#pragma unroll
    for (int r = 0; r < 8; ++r) {
      int m = rb + 16 * t + r + 8 * hg;
      qdst[(size_t)m * 32 + l15]      = (bf16)acc[t][0][0][r];
      qdst[(size_t)m * 32 + 16 + l15] = (bf16)acc[t][0][1][r];
      kdst[(size_t)m * 32 + l15]      = (bf16)acc[t][1][0][r];
      kdst[(size_t)m * 32 + 16 + l15] = (bf16)acc[t][1][1][r];
    }
  }
  // v transposed: contiguous 8 rows per lane -> single 16B store each.
  bf16* vdst = vT + (size_t)h * 32 * S_DIM;
#pragma unroll
  for (int t = 0; t < 2; ++t) {
    bf16x8 pack0, pack1;
#pragma unroll
    for (int r = 0; r < 8; ++r) {
      pack0[r] = (bf16)acc[t][2][0][r];
      pack1[r] = (bf16)acc[t][2][1][r];
    }
    *(bf16x8*)(vdst + (size_t)(l15) * S_DIM + rb + 16 * t + 8 * hg) = pack0;
    *(bf16x8*)(vdst + (size_t)(16 + l15) * S_DIM + rb + 16 * t + 8 * hg) = pack1;
  }
}

// ---------------------------------------------------------------------------
// Flash attention (causal via the reference's min(+/-1e5) mask semantics).
// One wave per (16-row tile, head); key chunks of 32; online softmax.
// ---------------------------------------------------------------------------
__global__ __launch_bounds__(32) void flash_kernel(
    const bf16* __restrict__ qh,  // [H, S, 32]
    const bf16* __restrict__ kh,  // [H, S, 32]
    const bf16* __restrict__ vT,  // [H, 32, S]
    bf16* __restrict__ O) {       // [S, H*32]
  const float SCALE = 1.0f / 256.0f;
  __shared__ __align__(16) bf16 pbuf[16 * 32];

  const int lane = threadIdx.x;
  const int l15 = lane & 15;
  const int hg = lane >> 4;
  const int it = blockIdx.x;
  const int h = blockIdx.y;
  const int rb = it * 16;

  const bf16* qbase = qh + (size_t)h * S_DIM * 32;
  const bf16* kbase = kh + (size_t)h * S_DIM * 32;
  const bf16* vbase = vT + (size_t)h * 32 * S_DIM;

  ABfrag aq;
  {
    const bf16* ar = qbase + (size_t)(rb + l15) * 32 + hg * 16;
    aq.h[0] = *(const bf16x8*)(ar);
    aq.h[1] = *(const bf16x8*)(ar + 8);
  }

  f32x8 o0 = {}, o1 = {};
  float mrun[8], lrun[8];
#pragma unroll
  for (int r = 0; r < 8; ++r) { mrun[r] = -3.0e38f; lrun[r] = 0.0f; }

  const int nch = (16 * it + 47) >> 5;  // causal: keys 0 .. rb+15
  for (int c = 0; c < nch; ++c) {
    const int kb = c * 32;
    f32x8 zero = {};
    ABfrag bk;
    {
      const bf16* kr = kbase + (size_t)(kb + l15) * 32 + hg * 16;
      bk.h[0] = *(const bf16x8*)kr;
      bk.h[1] = *(const bf16x8*)(kr + 8);
    }
    f32x8 d0 = WMMA_BF16(aq.v, bk.v, zero);
    {
      const bf16* kr = kbase + (size_t)(kb + 16 + l15) * 32 + hg * 16;
      bk.h[0] = *(const bf16x8*)kr;
      bk.h[1] = *(const bf16x8*)(kr + 8);
    }
    f32x8 d1 = WMMA_BF16(aq.v, bk.v, zero);

    float p0[8], p1[8], corr[8];
#pragma unroll
    for (int r = 0; r < 8; ++r) {
      const int row = rb + r + 8 * hg;
      const int c0 = kb + l15;
      const int c1 = kb + 16 + l15;
      float s0 = d0[r] * SCALE;
      s0 = (c0 <= row) ? fminf(s0, 1.0e5f) : -1.0e5f;
      float s1 = d1[r] * SCALE;
      s1 = (c1 <= row) ? fminf(s1, 1.0e5f) : -1.0e5f;
      float mx = fmaxf(s0, s1);
      mx = fmaxf(mx, __shfl_xor(mx, 1, 16));
      mx = fmaxf(mx, __shfl_xor(mx, 2, 16));
      mx = fmaxf(mx, __shfl_xor(mx, 4, 16));
      mx = fmaxf(mx, __shfl_xor(mx, 8, 16));
      const float mnew = fmaxf(mrun[r], mx);
      corr[r] = __expf(mrun[r] - mnew);
      p0[r] = __expf(s0 - mnew);
      p1[r] = __expf(s1 - mnew);
      float ls = p0[r] + p1[r];
      ls += __shfl_xor(ls, 1, 16);
      ls += __shfl_xor(ls, 2, 16);
      ls += __shfl_xor(ls, 4, 16);
      ls += __shfl_xor(ls, 8, 16);
      lrun[r] = lrun[r] * corr[r] + ls;
      mrun[r] = mnew;
    }

    // Transpose P (D-layout -> A-layout) through LDS as bf16.
#pragma unroll
    for (int r = 0; r < 8; ++r) {
      const int rl = r + 8 * hg;
      pbuf[rl * 32 + l15] = (bf16)p0[r];
      pbuf[rl * 32 + 16 + l15] = (bf16)p1[r];
    }
    asm volatile("s_wait_dscnt 0x0" ::: "memory");
    ABfrag ap;
    {
      const bf16* pr = &pbuf[l15 * 32 + hg * 16];
      ap.h[0] = *(const bf16x8*)pr;
      ap.h[1] = *(const bf16x8*)(pr + 8);
    }
    asm volatile("s_wait_dscnt 0x0" ::: "memory");

#pragma unroll
    for (int r = 0; r < 8; ++r) { o0[r] *= corr[r]; o1[r] *= corr[r]; }

    ABfrag bv;
    {
      const bf16* vr = vbase + (size_t)(l15) * S_DIM + kb + hg * 16;
      bv.h[0] = *(const bf16x8*)vr;
      bv.h[1] = *(const bf16x8*)(vr + 8);
    }
    o0 = WMMA_BF16(ap.v, bv.v, o0);
    {
      const bf16* vr = vbase + (size_t)(16 + l15) * S_DIM + kb + hg * 16;
      bv.h[0] = *(const bf16x8*)vr;
      bv.h[1] = *(const bf16x8*)(vr + 8);
    }
    o1 = WMMA_BF16(ap.v, bv.v, o1);
  }

#pragma unroll
  for (int r = 0; r < 8; ++r) {
    const int m = rb + r + 8 * hg;
    const float inv = 1.0f / lrun[r];
    O[(size_t)m * (H_DIM * 32) + h * 32 + l15] = (bf16)(o0[r] * inv);
    O[(size_t)m * (H_DIM * 32) + h * 32 + 16 + l15] = (bf16)(o1[r] * inv);
  }
}

// ---------------------------------------------------------------------------
// Generic bf16 WMMA GEMM: C[M,N] = A[M,K] @ B[K,N] (+bias +residual, relu)
// B supplied transposed as BT[N,K].  One wave -> 32x64 tile (2x4 WMMA block):
// 8 accumulators; A frags reused across 4 B columns, B frags across 2 A rows.
// ---------------------------------------------------------------------------
__global__ __launch_bounds__(32) void gemm_bf16_kernel(
    const bf16* __restrict__ A, const bf16* __restrict__ BT,
    const float* __restrict__ bias, const float* __restrict__ residual,
    float* __restrict__ outF, bf16* __restrict__ outBF,
    int M, int N, int K, int relu) {
  const int lane = threadIdx.x;
  const int l15 = lane & 15;
  const int hg = lane >> 4;
  const int nb = blockIdx.x * 64;
  const int rb = blockIdx.y * 32;

  const bf16* arow0 = A + (size_t)(rb + l15) * K + hg * 16;
  const bf16* arow1 = A + (size_t)(rb + 16 + l15) * K + hg * 16;

  f32x8 acc[2][4];
#pragma unroll
  for (int t = 0; t < 2; ++t)
#pragma unroll
    for (int j = 0; j < 4; ++j) acc[t][j] = (f32x8){};

  for (int kb = 0; kb < K; kb += 32) {
    ABfrag a0, a1;
    a0.h[0] = *(const bf16x8*)(arow0 + kb);
    a0.h[1] = *(const bf16x8*)(arow0 + kb + 8);
    a1.h[0] = *(const bf16x8*)(arow1 + kb);
    a1.h[1] = *(const bf16x8*)(arow1 + kb + 8);
#pragma unroll
    for (int j = 0; j < 4; ++j) {
      ABfrag b;
      const bf16* bc = BT + (size_t)(nb + j * 16 + l15) * K + hg * 16 + kb;
      b.h[0] = *(const bf16x8*)bc;
      b.h[1] = *(const bf16x8*)(bc + 8);
      acc[0][j] = WMMA_BF16(a0.v, b.v, acc[0][j]);
      acc[1][j] = WMMA_BF16(a1.v, b.v, acc[1][j]);
    }
  }

#pragma unroll
  for (int j = 0; j < 4; ++j) {
    const int n = nb + j * 16 + l15;
    const float bval = bias ? bias[n] : 0.0f;
#pragma unroll
    for (int t = 0; t < 2; ++t) {
#pragma unroll
      for (int r = 0; r < 8; ++r) {
        const int m = rb + 16 * t + r + 8 * hg;
        float v = acc[t][j][r] + bval;
        if (residual) v += residual[(size_t)m * N + n];
        if (relu) v = fmaxf(v, 0.0f);
        if (outF) outF[(size_t)m * N + n] = v;
        if (outBF) outBF[(size_t)m * N + n] = (bf16)v;
      }
    }
  }
}

// ---------------------------------------------------------------------------
// Row LayerNorm (no affine): out = (x - mean) / sqrt(var + eps)
// ---------------------------------------------------------------------------
__global__ __launch_bounds__(256) void ln_kernel(const float* __restrict__ in,
                                                 float* __restrict__ outF,
                                                 bf16* __restrict__ outBF,
                                                 int D) {
  __shared__ float s_sum[256];
  __shared__ float s_sq[256];
  const int row = blockIdx.x;
  const int tid = threadIdx.x;
  const float* p = in + (size_t)row * D;

  float sum = 0.0f, sq = 0.0f;
  for (int j = tid; j < D; j += 256) {
    float v = p[j];
    sum += v;
    sq += v * v;
  }
  s_sum[tid] = sum;
  s_sq[tid] = sq;
  __syncthreads();
  for (int s = 128; s > 0; s >>= 1) {
    if (tid < s) {
      s_sum[tid] += s_sum[tid + s];
      s_sq[tid] += s_sq[tid + s];
    }
    __syncthreads();
  }
  const float mean = s_sum[0] / D;
  const float var = s_sq[0] / D - mean * mean;
  const float rstd = rsqrtf(var + 1e-5f);

  for (int j = tid; j < D; j += 256) {
    float o = (p[j] - mean) * rstd;
    outF[(size_t)row * D + j] = o;
    if (outBF) outBF[(size_t)row * D + j] = (bf16)o;
  }
}

// ---------------------------------------------------------------------------
extern "C" void kernel_launch(void* const* d_in, const int* in_sizes, int n_in,
                              void* d_out, int out_size, void* d_ws,
                              size_t ws_size, hipStream_t stream) {
  (void)in_sizes; (void)n_in; (void)out_size; (void)ws_size;
  const float* x  = (const float*)d_in[0];
  const float* wq = (const float*)d_in[1];
  const float* wk = (const float*)d_in[2];
  const float* wv = (const float*)d_in[3];
  const float* W1 = (const float*)d_in[4];
  const float* b1 = (const float*)d_in[5];
  const float* W2 = (const float*)d_in[6];
  const float* b2 = (const float*)d_in[7];
  const float* W3 = (const float*)d_in[8];
  const float* b3 = (const float*)d_in[9];

  char* ws = (char*)d_ws;
  size_t off = 0;
  auto alloc = [&](size_t bytes) -> void* {
    void* p = ws + off;
    off += (bytes + 255) & ~(size_t)255;
    return p;
  };

  const int S = S_DIM, D = D_DIM, H = H_DIM;
  bf16* xh  = (bf16*)alloc((size_t)S * D * 2);
  bf16* wqT = (bf16*)alloc((size_t)H * 32 * D * 2);
  bf16* wkT = (bf16*)alloc((size_t)H * 32 * D * 2);
  bf16* wvT = (bf16*)alloc((size_t)H * 32 * D * 2);
  bf16* W1T = (bf16*)alloc((size_t)D * 128 * 2);
  bf16* W2T = (bf16*)alloc((size_t)FFN * D * 2);
  bf16* W3T = (bf16*)alloc((size_t)D * FFN * 2);
  bf16* qh  = (bf16*)alloc((size_t)H * S * 32 * 2);
  bf16* kh  = (bf16*)alloc((size_t)H * S * 32 * 2);
  bf16* vTb = (bf16*)alloc((size_t)H * 32 * S * 2);
  bf16* Ob  = (bf16*)alloc((size_t)S * 128 * 2);
  float* t1 = (float*)alloc((size_t)S * D * 4);   // reused for pre-LN2 too
  float* yb = (float*)alloc((size_t)S * D * 4);
  bf16* yh  = (bf16*)alloc((size_t)S * D * 2);
  bf16* zh  = (bf16*)alloc((size_t)S * FFN * 2);

  // 1) convert / transpose weights & activations to bf16
  {
    int n = S * D;
    convert_bf16_kernel<<<(n + 255) / 256, 256, 0, stream>>>(x, xh, n);
  }
  {
    int n = H * D * 32;
    transpose_bf16_kernel<<<(n + 255) / 256, 256, 0, stream>>>(wq, wqT, H, D, 32);
    transpose_bf16_kernel<<<(n + 255) / 256, 256, 0, stream>>>(wk, wkT, H, D, 32);
    transpose_bf16_kernel<<<(n + 255) / 256, 256, 0, stream>>>(wv, wvT, H, D, 32);
  }
  {
    int n = 128 * D;
    transpose_bf16_kernel<<<(n + 255) / 256, 256, 0, stream>>>(W1, W1T, 1, 128, D);
  }
  {
    int n = D * FFN;
    transpose_bf16_kernel<<<(n + 255) / 256, 256, 0, stream>>>(W2, W2T, 1, D, FFN);
    transpose_bf16_kernel<<<(n + 255) / 256, 256, 0, stream>>>(W3, W3T, 1, FFN, D);
  }

  // 2) QKV projection (WMMA, 32-row tiles)
  qkv_kernel<<<dim3(S / 32, H), 32, 0, stream>>>(xh, wqT, wkT, wvT, qh, kh, vTb);

  // 3) flash attention (WMMA + online softmax)
  flash_kernel<<<dim3(S / 16, H), 32, 0, stream>>>(qh, kh, vTb, Ob);

  // 4) output projection + residual(x):  t1 = x + O @ W1 + b1
  gemm_bf16_kernel<<<dim3(D / 64, S / 32), 32, 0, stream>>>(
      Ob, W1T, b1, x, t1, (bf16*)nullptr, S, D, 128, 0);

  // 5) LayerNorm 1 -> y (f32) + yh (bf16)
  ln_kernel<<<S, 256, 0, stream>>>(t1, yb, yh, D);

  // 6) FFN up + ReLU:  zh = relu(y @ W2 + b2) (bf16)
  gemm_bf16_kernel<<<dim3(FFN / 64, S / 32), 32, 0, stream>>>(
      yh, W2T, b2, (const float*)nullptr, (float*)nullptr, zh, S, FFN, D, 1);

  // 7) FFN down + residual(y):  t1 = y + z @ W3 + b3
  gemm_bf16_kernel<<<dim3(D / 64, S / 32), 32, 0, stream>>>(
      zh, W3T, b3, yb, t1, (bf16*)nullptr, S, D, FFN, 0);

  // 8) LayerNorm 2 -> output (f32)
  ln_kernel<<<S, 256, 0, stream>>>(t1, (float*)d_out, (bf16*)nullptr, D);
}
